// Custom_35502199669122
// MI455X (gfx1250) — compile-verified, matrix operand order checked by reading, not run on hardware
//
#include <hip/hip_runtime.h>
#include <hip/hip_bf16.h>

typedef __bf16 bf16;
typedef __attribute__((ext_vector_type(16))) __bf16 v16bf;
typedef __attribute__((ext_vector_type(8)))  float  v8f;
typedef __attribute__((ext_vector_type(4)))  unsigned int v4u;
typedef __attribute__((ext_vector_type(4)))  int v4i;

#define NPTS   262144     // B * NB
#define NBATCH 4
#define NBVOX  65536
#define CDIM   256
#define TOPK   4096
#define FPSP   2048

#define GROWS  256        // block row tile (2 x 16-row tiles per wave)
#define APAD   40         // A row stride (bf16): 80B -> conflict-free, 16B aligned
#define BPAD   264        // B col stride (bf16): 528B -> conflict-free, 16B aligned
#define GEMM_LDS_BYTES ((2 * GROWS * APAD + 64 * BPAD) * 2)

// CDNA5 async global->LDS path (guarded; falls back to load+ds_store).
// Probe-confirmed signature: (global v4i*, local v4i*, imm int offset, imm int cpol)
#if defined(__AMDGCN__) && __has_builtin(__builtin_amdgcn_global_load_async_to_lds_b128)
#define HAVE_ASYNC_LDS 1
typedef __attribute__((address_space(1))) v4i gv4i;
typedef __attribute__((address_space(3))) v4i lv4i;
#else
#define HAVE_ASYNC_LDS 0
#endif

static __device__ inline void wait_async_zero() {
#if HAVE_ASYNC_LDS
#if __has_builtin(__builtin_amdgcn_s_wait_asynccnt)
    __builtin_amdgcn_s_wait_asynccnt(0);
#else
    asm volatile("s_wait_asynccnt 0x0" ::: "memory");
#endif
#endif
}

// 16-bf16 fragment viewed as two 16-byte quads (matches WMMA A/B per-lane layout:
// two contiguous 8-element K runs at k = hi*8 and k = 16 + hi*8)
union Frag16 {
    v4u   q[2];
    v16bf v;
};

// ---------------------------------------------------------------------------
// fp32 -> bf16 conversion (grid-stride)
// ---------------------------------------------------------------------------
__global__ __launch_bounds__(256)
void f32_to_bf16_kernel(const float* __restrict__ src, bf16* __restrict__ dst, size_t n) {
    size_t i = (size_t)blockIdx.x * blockDim.x + threadIdx.x;
    size_t stride = (size_t)gridDim.x * blockDim.x;
    for (; i < n; i += stride) dst[i] = (bf16)src[i];
}

// ---------------------------------------------------------------------------
// Weight convert + transpose: W[k][n] f32 (256x256) -> Wt[n][k] bf16
// ---------------------------------------------------------------------------
__global__ __launch_bounds__(256)
void convert_transpose_w_kernel(const float* __restrict__ W, bf16* __restrict__ Wt) {
    int n = blockIdx.x;   // 256
    int k = threadIdx.x;  // 256
    Wt[n * CDIM + k] = (bf16)W[k * CDIM + n];
}

// ---------------------------------------------------------------------------
// GEMM: H(N x 256) = X(N x 256, bf16) @ W(256 x 256), f32 accum, WMMA bf16.
// Block: 256 threads = 8 waves; block tile 256 rows x 64 cols.
// Each wave: TWO 16-row tiles x 4 col-tiles (B fragments reused) -> 8 WMMA
// per K chunk for 12 ds_load_b128. A chunk double-buffered in LDS, staged via
// async global->LDS (ASYNCcnt) when available; B resident in LDS.
// ---------------------------------------------------------------------------
__global__ __launch_bounds__(256)
void gemm_bf16_wmma(const bf16* __restrict__ X, const bf16* __restrict__ Wt,
                    float* __restrict__ H, int N) {
    extern __shared__ __align__(16) bf16 smem[];
    bf16* As0 = smem;                       // GROWS*APAD
    bf16* As1 = smem + GROWS * APAD;        // GROWS*APAD
    bf16* Bs  = smem + 2 * GROWS * APAD;    // 64*BPAD, [c][k] padded

    const int tid  = threadIdx.x;
    const int wave = tid >> 5;
    const int lane = tid & 31;
    const int r    = lane & 15;
    const int hi   = lane >> 4;
    const int rowBase = blockIdx.x * GROWS;
    const int colBase = blockIdx.y * 64;

    // Preload B block: Wt rows [colBase, colBase+64) are contiguous -> coalesced
    for (int i = tid; i < 64 * 32; i += 256) {      // 2048 quads of 8 bf16
        int c = i >> 5;
        int j = (i & 31) * 8;
        *(v4u*)(Bs + c * BPAD + j) =
            *(const v4u*)(Wt + (size_t)(colBase + c) * CDIM + j);
    }

    // One thread per row: stage 32 bf16 (64B) per chunk
    const bf16* Xrow = X + (size_t)(rowBase + tid) * CDIM;

    auto stageA = [&](int kc, bf16* buf) {
        const bf16* src = Xrow + kc * 32;
        bf16* dst = buf + tid * APAD;
#if HAVE_ASYNC_LDS
        __builtin_amdgcn_global_load_async_to_lds_b128(
            (gv4i*)(src),      (lv4i*)(dst),      0, 0);
        __builtin_amdgcn_global_load_async_to_lds_b128(
            (gv4i*)(src + 8),  (lv4i*)(dst + 8),  0, 0);
        __builtin_amdgcn_global_load_async_to_lds_b128(
            (gv4i*)(src + 16), (lv4i*)(dst + 16), 0, 0);
        __builtin_amdgcn_global_load_async_to_lds_b128(
            (gv4i*)(src + 24), (lv4i*)(dst + 24), 0, 0);
#else
        const v4u* s = (const v4u*)src;
        v4u* d = (v4u*)dst;
        v4u t0 = s[0], t1 = s[1], t2 = s[2], t3 = s[3];
        d[0] = t0; d[1] = t1; d[2] = t2; d[3] = t3;
        __builtin_prefetch(src + 32, 0, 1);
#endif
    };

    stageA(0, As0);

    v8f acc[2][4];
    #pragma unroll
    for (int t = 0; t < 2; ++t)
        #pragma unroll
        for (int i = 0; i < 4; ++i)
            acc[t][i] = (v8f){0.f,0.f,0.f,0.f,0.f,0.f,0.f,0.f};

    wait_async_zero();
    __syncthreads();

    #pragma unroll
    for (int kc = 0; kc < 8; ++kc) {
        const int k0 = kc * 32;
        bf16* Acur = (kc & 1) ? As1 : As0;

        // Kick off next chunk's HBM->LDS transfer (overlaps with WMMA)
        if (kc < 7) stageA(kc + 1, (kc & 1) ? As0 : As1);

        // Two A fragments (row tiles wave*16 and 128 + wave*16): 2 b128 each
        const bf16* ar0 = Acur + (wave * 16 + r) * APAD;
        const bf16* ar1 = Acur + (128 + wave * 16 + r) * APAD;
        Frag16 a0, a1;
        a0.q[0] = *(const v4u*)(ar0 + hi * 8);
        a0.q[1] = *(const v4u*)(ar0 + 16 + hi * 8);
        a1.q[0] = *(const v4u*)(ar1 + hi * 8);
        a1.q[1] = *(const v4u*)(ar1 + 16 + hi * 8);

        #pragma unroll
        for (int ct = 0; ct < 4; ++ct) {
            const bf16* br = Bs + (ct * 16 + r) * BPAD + k0;
            Frag16 b;
            b.q[0] = *(const v4u*)(br + hi * 8);
            b.q[1] = *(const v4u*)(br + 16 + hi * 8);
            acc[0][ct] = __builtin_amdgcn_wmma_f32_16x16x32_bf16(
                false, a0.v, false, b.v, (short)0, acc[0][ct], false, false);
            acc[1][ct] = __builtin_amdgcn_wmma_f32_16x16x32_bf16(
                false, a1.v, false, b.v, (short)0, acc[1][ct], false, false);
        }

        wait_async_zero();
        __syncthreads();
    }

    // C/D layout: VGPR v, lanes 0-15 -> (M=v, N=lane); lanes 16-31 -> (M=v+8, N=lane-16)
    #pragma unroll
    for (int t = 0; t < 2; ++t) {
        #pragma unroll
        for (int ct = 0; ct < 4; ++ct) {
            #pragma unroll
            for (int v = 0; v < 8; ++v) {
                int row = rowBase + t * 128 + wave * 16 + v + hi * 8;
                int col = colBase + ct * 16 + r;
                H[(size_t)row * CDIM + col] = acc[t][ct][v];
            }
        }
    }
}

// ---------------------------------------------------------------------------
// BN batch statistics: per-column sum / sumsq over N rows
// ---------------------------------------------------------------------------
__global__ __launch_bounds__(256)
void bn_stats_kernel(const float* __restrict__ H, float* __restrict__ stats, int N) {
    int c = threadIdx.x;
    float s = 0.f, q = 0.f;
    for (int r = blockIdx.x; r < N; r += gridDim.x) {
        float h = H[(size_t)r * CDIM + c];
        s += h; q += h * h;
    }
    atomicAdd(&stats[c], s);
    atomicAdd(&stats[CDIM + c], q);
}

__global__ void bn_finalize_kernel(const float* __restrict__ stats,
                                   const float* __restrict__ gamma,
                                   const float* __restrict__ beta,
                                   float* __restrict__ ss, float invN) {
    int c = threadIdx.x;
    float mean = stats[c] * invN;
    float var  = stats[CDIM + c] * invN - mean * mean;
    float sc   = rsqrtf(var + 1e-5f) * gamma[c];
    ss[c]        = sc;
    ss[CDIM + c] = beta[c] - mean * sc;
}

// ---------------------------------------------------------------------------
// Layer 2: y = relu(bn(h)) @ W2(256x3) + b2, one wave per row.
// mode 0 (offset): clip -> votes(N,4) / seeds(N,4)
// mode 1 (cls):    seed_cls(N,3) + score = max over classes
// ---------------------------------------------------------------------------
__global__ __launch_bounds__(256)
void layer2_kernel(const float* __restrict__ H, const float* __restrict__ ss,
                   const float* __restrict__ w2, const float* __restrict__ b2,
                   const float* __restrict__ seed_xyz,
                   float* __restrict__ out_a, float* __restrict__ out_b,
                   int mode, int N) {
    int wave = threadIdx.x >> 5;
    int lane = threadIdx.x & 31;
    int row  = blockIdx.x * 8 + wave;
    if (row >= N) return;

    float p0 = 0.f, p1 = 0.f, p2 = 0.f;
    #pragma unroll
    for (int j = 0; j < 8; ++j) {
        int c = lane + 32 * j;
        float h = H[(size_t)row * CDIM + c];
        float y = fmaxf(h * ss[c] + ss[CDIM + c], 0.f);
        p0 += y * w2[c * 3 + 0];
        p1 += y * w2[c * 3 + 1];
        p2 += y * w2[c * 3 + 2];
    }
    #pragma unroll
    for (int off = 16; off; off >>= 1) {
        p0 += __shfl_down(p0, off, 32);
        p1 += __shfl_down(p1, off, 32);
        p2 += __shfl_down(p2, off, 32);
    }
    if (lane == 0) {
        p0 += b2[0]; p1 += b2[1]; p2 += b2[2];
        if (mode == 0) {
            float x = seed_xyz[(size_t)row * 3 + 0];
            float y = seed_xyz[(size_t)row * 3 + 1];
            float z = seed_xyz[(size_t)row * 3 + 2];
            float vx = x + fminf(fmaxf(p0, -3.f), 3.f);
            float vy = y + fminf(fmaxf(p1, -3.f), 3.f);
            float vz = z + fminf(fmaxf(p2, -2.f), 2.f);
            float bc = (float)(row >> 16);  // row / NBVOX
            out_a[(size_t)row * 4 + 0] = bc;   // votes
            out_a[(size_t)row * 4 + 1] = vx;
            out_a[(size_t)row * 4 + 2] = vy;
            out_a[(size_t)row * 4 + 3] = vz;
            out_b[(size_t)row * 4 + 0] = bc;   // seeds
            out_b[(size_t)row * 4 + 1] = x;
            out_b[(size_t)row * 4 + 2] = y;
            out_b[(size_t)row * 4 + 3] = z;
        } else {
            out_a[(size_t)row * 3 + 0] = p0;   // seed_cls
            out_a[(size_t)row * 3 + 1] = p1;
            out_a[(size_t)row * 3 + 2] = p2;
            out_b[row] = fmaxf(p0, fmaxf(p1, p2));  // score (sigmoid is monotone)
        }
    }
}

// ---------------------------------------------------------------------------
// Per-batch top-K (K=4096 of 65536) via 4096-bin histogram select.
// ---------------------------------------------------------------------------
__global__ __launch_bounds__(1024)
void topk_select_kernel(const float* __restrict__ scores,
                        const float* __restrict__ seed_xyz,
                        int* __restrict__ topk_idx, float* __restrict__ xyz_top) {
    __shared__ unsigned hist[4096];
    __shared__ unsigned s_th, s_G, cntHi, cntEq;
    const int b = blockIdx.x;
    const float* sc = scores + (size_t)b * NBVOX;

    for (int i = threadIdx.x; i < 4096; i += 1024) hist[i] = 0;
    __syncthreads();

    for (int i = threadIdx.x; i < NBVOX; i += 1024) {
        unsigned u = __float_as_uint(sc[i]);
        u = (u & 0x80000000u) ? ~u : (u | 0x80000000u);  // order-preserving key
        atomicAdd(&hist[u >> 20], 1u);
    }
    __syncthreads();

    if (threadIdx.x == 0) {
        unsigned acc = 0;
        for (int bin = 4095; bin >= 0; --bin) {
            unsigned nacc = acc + hist[bin];
            if (nacc >= (unsigned)TOPK) { s_th = bin; s_G = acc; break; }
            acc = nacc;
        }
        cntHi = 0; cntEq = 0;
    }
    __syncthreads();

    const unsigned th = s_th, G = s_G;
    for (int i = threadIdx.x; i < NBVOX; i += 1024) {
        unsigned u = __float_as_uint(sc[i]);
        u = (u & 0x80000000u) ? ~u : (u | 0x80000000u);
        unsigned bin = u >> 20;
        int slot = -1;
        if (bin > th) {
            slot = (int)atomicAdd(&cntHi, 1u);
        } else if (bin == th) {
            unsigned e = atomicAdd(&cntEq, 1u);
            if (G + e < (unsigned)TOPK) slot = (int)(G + e);
        }
        if (slot >= 0) {
            topk_idx[b * TOPK + slot] = i;
            const float* p = seed_xyz + ((size_t)b * NBVOX + i) * 3;
            float* q = xyz_top + ((size_t)b * TOPK + slot) * 3;
            q[0] = p[0]; q[1] = p[1]; q[2] = p[2];
        }
    }
}

// ---------------------------------------------------------------------------
// D-FPS: 2048 serial iterations over 4096 points. One block per batch.
// ---------------------------------------------------------------------------
__global__ __launch_bounds__(1024)
void fps_kernel(const float* __restrict__ xyz_top, int* __restrict__ sel) {
    __shared__ float dist[TOPK];         // 16 KB
    __shared__ float redv[32];
    __shared__ int   redi[32];
    __shared__ float lastx, lasty, lastz;
    const int b = blockIdx.x;
    const float* xyz = xyz_top + (size_t)b * TOPK * 3;
    const int t = threadIdx.x;

    for (int i = t; i < TOPK; i += 1024) dist[i] = 1e10f;
    if (t == 0) {
        sel[b * FPSP + 0] = 0;
        lastx = xyz[0]; lasty = xyz[1]; lastz = xyz[2];
    }
    __syncthreads();

    for (int p = 1; p < FPSP; ++p) {
        const float lx = lastx, ly = lasty, lz = lastz;
        float bestv = -1.f; int besti = 0;
        #pragma unroll
        for (int j = 0; j < 4; ++j) {
            int i = t + j * 1024;
            float dx = xyz[i * 3 + 0] - lx;
            float dy = xyz[i * 3 + 1] - ly;
            float dz = xyz[i * 3 + 2] - lz;
            float d = dx * dx + dy * dy + dz * dz;
            float m = fminf(dist[i], d);
            dist[i] = m;
            if (m > bestv) { bestv = m; besti = i; }
        }
        #pragma unroll
        for (int off = 16; off; off >>= 1) {
            float ov = __shfl_down(bestv, off, 32);
            int   oi = __shfl_down(besti, off, 32);
            if (ov > bestv) { bestv = ov; besti = oi; }
        }
        if ((t & 31) == 0) { redv[t >> 5] = bestv; redi[t >> 5] = besti; }
        __syncthreads();
        if (t < 32) {
            bestv = redv[t]; besti = redi[t];
            #pragma unroll
            for (int off = 16; off; off >>= 1) {
                float ov = __shfl_down(bestv, off, 32);
                int   oi = __shfl_down(besti, off, 32);
                if (ov > bestv) { bestv = ov; besti = oi; }
            }
            if (t == 0) {
                sel[b * FPSP + p] = besti;
                lastx = xyz[besti * 3 + 0];
                lasty = xyz[besti * 3 + 1];
                lastz = xyz[besti * 3 + 2];
            }
        }
        __syncthreads();
    }
}

// ---------------------------------------------------------------------------
// Final gather: fps local idx -> topk idx -> global idx
// ---------------------------------------------------------------------------
__global__ __launch_bounds__(256)
void gather_kernel(const int* __restrict__ topk_idx, const int* __restrict__ sel,
                   const float* __restrict__ seed_feats, const float* __restrict__ votes,
                   float* __restrict__ cand, float* __restrict__ feat,
                   float* __restrict__ fidx) {
    const int row = blockIdx.x;          // 0 .. B*P-1
    const int b = row >> 11;             // / FPSP
    const int p = row & (FPSP - 1);
    const int fi = sel[b * FPSP + p];
    const int ti = topk_idx[b * TOPK + fi];
    const int g  = b * NBVOX + ti;
    const int c  = threadIdx.x;
    feat[(size_t)row * CDIM + c] = seed_feats[(size_t)g * CDIM + c];
    if (c < 4)  cand[(size_t)row * 4 + c] = votes[(size_t)g * 4 + c];
    if (c == 0) fidx[row] = (float)g;
}

// ---------------------------------------------------------------------------
extern "C" void kernel_launch(void* const* d_in, const int* in_sizes, int n_in,
                              void* d_out, int out_size, void* d_ws, size_t ws_size,
                              hipStream_t stream) {
    (void)in_sizes; (void)n_in; (void)out_size; (void)ws_size;
    const float* seed_xyz   = (const float*)d_in[0];
    const float* seed_feats = (const float*)d_in[1];
    const float* w_off1     = (const float*)d_in[2];
    const float* g_off1     = (const float*)d_in[3];
    const float* be_off1    = (const float*)d_in[4];
    const float* w_off2     = (const float*)d_in[5];
    const float* b_off2     = (const float*)d_in[6];
    const float* w_cls1     = (const float*)d_in[7];
    const float* g_cls1     = (const float*)d_in[8];
    const float* be_cls1    = (const float*)d_in[9];
    const float* w_cls2     = (const float*)d_in[10];
    const float* b_cls2     = (const float*)d_in[11];

    const int N = NPTS;

    // workspace carve-up
    char* ws = (char*)d_ws;
    size_t off = 0;
    auto carve = [&](size_t bytes) -> void* {
        void* p = ws + off;
        off += (bytes + 255) & ~(size_t)255;
        return p;
    };
    bf16*  Xbf     = (bf16*)carve((size_t)N * CDIM * 2);
    bf16*  Wt_off  = (bf16*)carve((size_t)CDIM * CDIM * 2);
    bf16*  Wt_cls  = (bf16*)carve((size_t)CDIM * CDIM * 2);
    float* H       = (float*)carve((size_t)N * CDIM * 4);
    float* stats   = (float*)carve(2 * CDIM * 4);
    float* ss      = (float*)carve(2 * CDIM * 4);
    float* scores  = (float*)carve((size_t)N * 4);
    int*   topkIdx = (int*)carve((size_t)NBATCH * TOPK * 4);
    float* xyzTop  = (float*)carve((size_t)NBATCH * TOPK * 3 * 4);
    int*   fpsSel  = (int*)carve((size_t)NBATCH * FPSP * 4);

    // output carve-up (return-order concatenation, all float)
    float* out     = (float*)d_out;
    float* o_cand  = out;
    float* o_feat  = o_cand + (size_t)NBATCH * FPSP * 4;
    float* o_seeds = o_feat + (size_t)NBATCH * FPSP * CDIM;
    float* o_cls   = o_seeds + (size_t)N * 4;
    float* o_votes = o_cls + (size_t)N * 3;
    float* o_fidx  = o_votes + (size_t)N * 4;

    // 1) conversions (weights transposed for contiguous B fragments)
    f32_to_bf16_kernel<<<2048, 256, 0, stream>>>(seed_feats, Xbf, (size_t)N * CDIM);
    convert_transpose_w_kernel<<<CDIM, CDIM, 0, stream>>>(w_off1, Wt_off);
    convert_transpose_w_kernel<<<CDIM, CDIM, 0, stream>>>(w_cls1, Wt_cls);

    dim3 ggrid(N / GROWS, CDIM / 64);

    // 2) offset MLP
    gemm_bf16_wmma<<<ggrid, 256, GEMM_LDS_BYTES, stream>>>(Xbf, Wt_off, H, N);
    (void)hipMemsetAsync(stats, 0, 2 * CDIM * 4, stream);
    bn_stats_kernel<<<1024, 256, 0, stream>>>(H, stats, N);
    bn_finalize_kernel<<<1, 256, 0, stream>>>(stats, g_off1, be_off1, ss, 1.f / (float)N);
    layer2_kernel<<<N / 8, 256, 0, stream>>>(H, ss, w_off2, b_off2, seed_xyz,
                                             o_votes, o_seeds, 0, N);

    // 3) cls MLP (reuses H)
    gemm_bf16_wmma<<<ggrid, 256, GEMM_LDS_BYTES, stream>>>(Xbf, Wt_cls, H, N);
    (void)hipMemsetAsync(stats, 0, 2 * CDIM * 4, stream);
    bn_stats_kernel<<<1024, 256, 0, stream>>>(H, stats, N);
    bn_finalize_kernel<<<1, 256, 0, stream>>>(stats, g_cls1, be_cls1, ss, 1.f / (float)N);
    layer2_kernel<<<N / 8, 256, 0, stream>>>(H, ss, w_cls2, b_cls2, seed_xyz,
                                             o_cls, scores, 1, N);

    // 4) per-batch top-k + FPS + gather
    topk_select_kernel<<<NBATCH, 1024, 0, stream>>>(scores, seed_xyz, topkIdx, xyzTop);
    fps_kernel<<<NBATCH, 1024, 0, stream>>>(xyzTop, fpsSel);
    gather_kernel<<<NBATCH * FPSP, 256, 0, stream>>>(topkIdx, fpsSel, seed_feats,
                                                     o_votes, o_cand, o_feat, o_fidx);
}
